// EGU_88897233092648
// MI455X (gfx1250) — compile-verified, matrix operand order checked by reading, not run on hardware
//
#include <hip/hip_runtime.h>
#include <hip/hip_bf16.h>

typedef __attribute__((ext_vector_type(16))) __bf16 v16bf;
typedef __attribute__((ext_vector_type(8)))  __bf16 v8bf;
typedef __attribute__((ext_vector_type(2)))  __bf16 v2bf;
typedef __attribute__((ext_vector_type(8)))  float  v8f;

#define T_STEPS 1024
#define BATCH   32
#define HDIM    1024
#define BH      (BATCH*HDIM)
#define TBH     (T_STEPS*BATCH*HDIM)
#define NBLK    64       // workgroups in persistent recurrence kernel
#define APAD    40       // 32 + 8 pad: 80B row stride, 16B aligned, bank-spread
#define WPAD    1032     // 1024 + 8 pad: 2064B row stride, 16B aligned

// Build a 16-element bf16 WMMA fragment from two contiguous 16-byte runs.
__device__ __forceinline__ v16bf ldfrag(const __bf16* p) {
    v8bf lo = *(const v8bf*)(p);
    v8bf hi = *(const v8bf*)(p + 16);
    return __builtin_shufflevector(lo, hi, 0, 1, 2, 3, 4, 5, 6, 7,
                                   8, 9, 10, 11, 12, 13, 14, 15);
}

// ---------------------------------------------------------------- grid barrier
__device__ __forceinline__ void gridBarrier(unsigned* cnt, unsigned* gen, unsigned nb) {
    __syncthreads();
    if (threadIdx.x == 0) {
        unsigned g = __hip_atomic_load(gen, __ATOMIC_ACQUIRE, __HIP_MEMORY_SCOPE_AGENT);
        unsigned a = __hip_atomic_fetch_add(cnt, 1u, __ATOMIC_ACQ_REL, __HIP_MEMORY_SCOPE_AGENT);
        if (a == nb - 1u) {
            __hip_atomic_store(cnt, 0u, __ATOMIC_RELAXED, __HIP_MEMORY_SCOPE_AGENT);
            __hip_atomic_fetch_add(gen, 1u, __ATOMIC_ACQ_REL, __HIP_MEMORY_SCOPE_AGENT);
        } else {
            while (__hip_atomic_load(gen, __ATOMIC_ACQUIRE, __HIP_MEMORY_SCOPE_AGENT) == g)
                __builtin_amdgcn_s_sleep(2);
        }
    }
    __syncthreads();
}

// --------------------------------------------------------- init h state + bars
__global__ __launch_bounds__(256) void init_kernel(const float* __restrict__ h0,
                                                   float* __restrict__ hf32,
                                                   __bf16* __restrict__ hbf,
                                                   unsigned* cnt, unsigned* gen) {
    int i = blockIdx.x * 256 + threadIdx.x;
    if (i < BH) {
        float v = h0[i];
        hf32[i] = v;
        hbf[i]  = (__bf16)v;
    }
    if (i == 0) { *cnt = 0u; *gen = 0u; }
}

// ------------------------------------------------------------- input projection
// Aout[m,n] = (bf16) tanh( sum_k X[m,k]*W[n,k] + bias[n] )
// block tile 128(M) x 64(N), 8 waves = 4x2, each wave 32x32 (2x2 WMMA tiles).
// Both LDS tiles are K-contiguous per row/column so fragments load as b128 pairs.
template <typename TA>
__global__ __launch_bounds__(256) void proj_kernel(const TA* __restrict__ Xsrc,
                                                   const float* __restrict__ W,
                                                   const float* __restrict__ bias,
                                                   __bf16* __restrict__ Aout) {
    __shared__ __align__(16) __bf16 As[128 * APAD];   // [m][k] 10 KB
    __shared__ __align__(16) __bf16 Bs[64 * APAD];    // [n][k]  5 KB (direct copy of W rows)
    const int tid  = threadIdx.x;
    const int lane = tid & 31;
    const int wv   = tid >> 5;
    const int wm   = wv >> 1;          // 0..3  (M group)
    const int wn   = wv & 1;           // 0..1  (N group)
    const int lh   = lane >> 4;        // K-half select per ISA 16-bit A/B layout
    const int l16  = lane & 15;
    const int m0   = blockIdx.y * 128;
    const int n0   = blockIdx.x * 64;

    v8f acc[2][2] = {};
    for (int kc = 0; kc < HDIM / 32; ++kc) {
        // stage A tile 128x32 -> bf16 pairs (one ds_store_b32 per pair)
        #pragma unroll
        for (int i = 0; i < 8; ++i) {
            int p = tid + i * 256;               // pair index, element e = 2p
            int r = p >> 4, c = (p & 15) * 2;
            const size_t g = (size_t)(m0 + r) * HDIM + kc * 32 + c;
            v2bf v; v[0] = (__bf16)(float)Xsrc[g]; v[1] = (__bf16)(float)Xsrc[g + 1];
            *(v2bf*)&As[r * APAD + c] = v;
        }
        // stage B tile 64x32 from W[N,K] row-major: direct copy, K-contiguous
        #pragma unroll
        for (int i = 0; i < 4; ++i) {
            int p = tid + i * 256;
            int n = p >> 4, c = (p & 15) * 2;
            const size_t g = (size_t)(n0 + n) * HDIM + kc * 32 + c;
            v2bf v; v[0] = (__bf16)W[g]; v[1] = (__bf16)W[g + 1];
            *(v2bf*)&Bs[n * APAD + c] = v;
        }
        if (kc + 1 < HDIM / 32)   // hint next A chunk into caches
            __builtin_prefetch(&Xsrc[(size_t)(m0 + (tid >> 1)) * HDIM + (kc + 1) * 32], 0, 1);
        __syncthreads();

        v16bf af[2], bfr[2];
        #pragma unroll
        for (int mt = 0; mt < 2; ++mt)
            af[mt] = ldfrag(&As[(wm * 32 + mt * 16 + l16) * APAD + lh * 8]);
        #pragma unroll
        for (int nt = 0; nt < 2; ++nt)
            bfr[nt] = ldfrag(&Bs[(wn * 32 + nt * 16 + l16) * APAD + lh * 8]);

        #pragma unroll
        for (int mt = 0; mt < 2; ++mt)
            #pragma unroll
            for (int nt = 0; nt < 2; ++nt)
                acc[mt][nt] = __builtin_amdgcn_wmma_f32_16x16x32_bf16(
                    false, af[mt], false, bfr[nt], (short)0, acc[mt][nt], false, false);
        __syncthreads();
    }

    // epilogue: bias + tanh, store bf16. C layout: lane=N, VGPR v = M row (+8 hi half)
    #pragma unroll
    for (int mt = 0; mt < 2; ++mt) {
        #pragma unroll
        for (int nt = 0; nt < 2; ++nt) {
            int ncol = n0 + wn * 32 + nt * 16 + l16;
            float bv = bias[ncol];
            #pragma unroll
            for (int v = 0; v < 8; ++v) {
                int mrow = m0 + wm * 32 + mt * 16 + lh * 8 + v;
                Aout[(size_t)mrow * HDIM + ncol] = (__bf16)tanhf(acc[mt][nt][v] + bv);
            }
        }
    }
}

// -------------------------------------------------------- persistent recurrence
// 64 workgroups, each owns a 16-wide output slab; w_hh slab resident in LDS
// (stored [n][k] so B fragments are two ds_load_b128).
// per step: z = h @ W^T + b ; g = sigmoid(z) ; h = g*h + (1-g)*tanh(a_t)
__global__ __launch_bounds__(256) void recur_kernel(
    const __bf16* __restrict__ At,   // [T][B][H] precomputed tanh(x@Wih^T+b), bf16
    const float*  __restrict__ Whh,  // [N=H][K=H] fp32 (original w_hh)
    const float*  __restrict__ bhh,
    __bf16* __restrict__ ybf,        // layer0 output for next layer's GEMM (or null)
    float*  __restrict__ yf32,       // layer1 output -> d_out x region (or null)
    float*  __restrict__ hy,         // final hidden -> d_out tail
    float*  __restrict__ hf32,       // [2][B][H] fp32 shadow
    __bf16* __restrict__ hbf,        // [2][B][H] bf16 GEMM operand
    unsigned* cnt, unsigned* gen) {
    __shared__ __align__(16) __bf16 Ws[16 * WPAD];   // resident slab ~32 KB, [n][k]
    __shared__ float Ps[8][BATCH][16];               // per-wave partials 16 KB
    const int tid  = threadIdx.x;
    const int lane = tid & 31;
    const int wv   = tid >> 5;                // wave owns K range [wv*128, wv*128+128)
    const int lh   = lane >> 4;
    const int l16  = lane & 15;
    const int n0   = blockIdx.x * 16;

    // stage weight slab once: Ws[n][k] = (bf16) Whh[n0+n][k]  (coalesced, paired)
    #pragma unroll
    for (int i = 0; i < 32; ++i) {
        int p = tid + i * 256;                // pair index over 16*1024/2
        int n = p >> 9, k = (p & 511) * 2;
        const size_t g = (size_t)(n0 + n) * HDIM + k;
        v2bf v; v[0] = (__bf16)Whh[g]; v[1] = (__bf16)Whh[g + 1];
        *(v2bf*)&Ws[n * WPAD + k] = v;
    }
    __syncthreads();

    int cur = 0;
    for (int t = 0; t < T_STEPS; ++t) {
        const __bf16* hsrc = hbf + (size_t)cur * BH;
        v8f acc0 = {}, acc1 = {};
        #pragma unroll
        for (int kk = 0; kk < 4; ++kk) {
            const int kc = wv * 4 + kk;
            v16bf bfr = ldfrag(&Ws[l16 * WPAD + kc * 32 + lh * 8]);
            v16bf af0 = ldfrag(&hsrc[(size_t)l16 * HDIM + kc * 32 + lh * 8]);
            v16bf af1 = ldfrag(&hsrc[(size_t)(16 + l16) * HDIM + kc * 32 + lh * 8]);
            acc0 = __builtin_amdgcn_wmma_f32_16x16x32_bf16(false, af0, false, bfr,
                                                           (short)0, acc0, false, false);
            acc1 = __builtin_amdgcn_wmma_f32_16x16x32_bf16(false, af1, false, bfr,
                                                           (short)0, acc1, false, false);
        }
        #pragma unroll
        for (int v = 0; v < 8; ++v) {
            Ps[wv][lh * 8 + v][l16]      = acc0[v];
            Ps[wv][16 + lh * 8 + v][l16] = acc1[v];
        }
        __syncthreads();

        const int nxt = cur ^ 1;
        #pragma unroll
        for (int q = 0; q < 2; ++q) {
            int idx = tid + q * 256;          // 512 outputs, 2 per thread
            int m = idx >> 4, n = idx & 15;
            float z = bhh[n0 + n];
            #pragma unroll
            for (int w = 0; w < 8; ++w) z += Ps[w][m][n];
            float g   = 1.f / (1.f + __expf(-z));
            size_t ho = (size_t)m * HDIM + n0 + n;
            float hold = hf32[(size_t)cur * BH + ho];
            float a    = (float)At[(size_t)t * BH + ho];
            float hn   = g * hold + (1.f - g) * a;
            hf32[(size_t)nxt * BH + ho] = hn;
            hbf[(size_t)nxt * BH + ho]  = (__bf16)hn;
            if (ybf)  ybf[(size_t)t * BH + ho]  = (__bf16)hn;
            if (yf32) yf32[(size_t)t * BH + ho] = hn;
            if (t == T_STEPS - 1) hy[ho] = hn;
        }
        gridBarrier(cnt, gen, (unsigned)gridDim.x);
        cur = nxt;
    }
}

// ------------------------------------------------------------------------ host
extern "C" void kernel_launch(void* const* d_in, const int* in_sizes, int n_in,
                              void* d_out, int out_size, void* d_ws, size_t ws_size,
                              hipStream_t stream) {
    const float* X    = (const float*)d_in[0];
    const float* hx   = (const float*)d_in[1];
    const float* wih0 = (const float*)d_in[2];
    const float* whh0 = (const float*)d_in[3];
    const float* bih0 = (const float*)d_in[4];
    const float* bhh0 = (const float*)d_in[5];
    const float* wih1 = (const float*)d_in[6];
    const float* whh1 = (const float*)d_in[7];
    const float* bih1 = (const float*)d_in[8];
    const float* bhh1 = (const float*)d_in[9];
    float* out = (float*)d_out;

    char* ws = (char*)d_ws;
    size_t off = 0;
    __bf16* A0  = (__bf16*)(ws + off); off += (size_t)TBH * 2;       // tanh proj L0
    __bf16* A1  = (__bf16*)(ws + off); off += (size_t)TBH * 2;       // tanh proj L1
    __bf16* Y0  = (__bf16*)(ws + off); off += (size_t)TBH * 2;       // layer0 output
    float*  HF  = (float*) (ws + off); off += (size_t)2 * BH * 4;    // h fp32 (x2)
    __bf16* HB  = (__bf16*)(ws + off); off += (size_t)2 * BH * 2;    // h bf16 (x2)
    unsigned* CNT = (unsigned*)(ws + off);
    unsigned* GEN = CNT + 1;

    const dim3 pgrid(HDIM / 64, (T_STEPS * BATCH) / 128);

    // ---- layer 0
    proj_kernel<float><<<pgrid, 256, 0, stream>>>(X, wih0, bih0, A0);
    init_kernel<<<BH / 256, 256, 0, stream>>>(hx, HF, HB, CNT, GEN);
    recur_kernel<<<NBLK, 256, 0, stream>>>(A0, whh0, bhh0, Y0, nullptr,
                                           out + (size_t)TBH, HF, HB, CNT, GEN);
    // ---- layer 1
    proj_kernel<__bf16><<<pgrid, 256, 0, stream>>>(Y0, wih1, bih1, A1);
    init_kernel<<<BH / 256, 256, 0, stream>>>(hx + BH, HF, HB, CNT, GEN);
    recur_kernel<<<NBLK, 256, 0, stream>>>(A1, whh1, bhh1, nullptr, out,
                                           out + (size_t)TBH + BH, HF, HB, CNT, GEN);
}
